// AttentionInPnts_83408264888938
// MI455X (gfx1250) — matrix operand order repaired and checked by reading, not compiled, not run on hardware
//
#include <hip/hip_runtime.h>

// ---------------- problem constants ----------------
#define BS    8
#define NPTS  4096
#define KNB   16
#define C     128
#define CH    64          // c/2
#define TP    8           // points per workgroup
#define NROW  (TP*KNB+16) // 144 rows: 128 near + 8 center + 8 zero pad
#define NRT   (NROW/16)   // 9 row tiles
#define NCH   256         // combined output channels: k(64) | v(128) | q(64)

// LDS strides (in elements) — padded for bank spread, dword aligned
#define ASTR  130
#define WSTR  130
#define KQSTR 68
#define VSTR  132

// LDS layout (bytes)
#define OFF_AHI 0
#define SZ_A    (NROW*ASTR*2)          // 37440
#define OFF_ALO (OFF_AHI + SZ_A)
#define OFF_WHI (OFF_ALO + SZ_A)       // 74880
#define SZ_W    (NCH*WSTR*2)           // 66560
#define OFF_WLO (OFF_WHI + SZ_W)
#define OFF_OUT (OFF_WLO + SZ_W)       // 208000 ; staging / k,q region / v region
#define SZ_OUT  (NROW*VSTR*4)          // 76032
#define OFF_QO  (OFF_OUT + NROW*KQSTR*4)
#define OFF_LOG (OFF_OUT + SZ_OUT)     // 284032
#define SMEM_BYTES (OFF_LOG + 2*TP*17*4)  // 285120 < 320KB

// CDNA5 async global->LDS DMA path (guarded: falls back to B128 register loads)
#if defined(__HIP_DEVICE_COMPILE__) && \
    __has_builtin(__builtin_amdgcn_global_load_async_to_lds_b128) && \
    __has_builtin(__builtin_amdgcn_s_wait_asynccnt)
#define HAVE_ASYNC_LDS 1
#else
#define HAVE_ASYNC_LDS 0
#endif

#if HAVE_ASYNC_LDS
typedef __attribute__((__vector_size__(16))) int v4i_t;
typedef __attribute__((address_space(1))) v4i_t* gv4i_p;   // global (AS1)
typedef __attribute__((address_space(3))) v4i_t* lv4i_p;   // LDS (AS3)
#endif

typedef __attribute__((ext_vector_type(16))) __bf16 v16bf;
typedef __attribute__((ext_vector_type(8)))  float  v8f;

union FragU { v16bf v; unsigned int u[8]; };

__device__ __forceinline__ unsigned short f2bf(float f) {
  union { float f; unsigned u; } v; v.f = f;
  unsigned r = v.u + 0x7FFFu + ((v.u >> 16) & 1u);   // round-to-nearest-even
  return (unsigned short)(r >> 16);
}
__device__ __forceinline__ float bf2f(unsigned short h) {
  union { float f; unsigned u; } v; v.u = ((unsigned)h) << 16; return v.f;
}

// split f32x4 -> bf16 hi/lo, packed dword stores (addresses are 4B aligned: k%4==0)
__device__ __forceinline__ void split_store(unsigned short* hi, unsigned short* lo, float4 v) {
  unsigned short h0 = f2bf(v.x), h1 = f2bf(v.y), h2 = f2bf(v.z), h3 = f2bf(v.w);
  unsigned short l0 = f2bf(v.x - bf2f(h0)), l1 = f2bf(v.y - bf2f(h1));
  unsigned short l2 = f2bf(v.z - bf2f(h2)), l3 = f2bf(v.w - bf2f(h3));
  ((unsigned*)hi)[0] = (unsigned)h0 | ((unsigned)h1 << 16);
  ((unsigned*)hi)[1] = (unsigned)h2 | ((unsigned)h3 << 16);
  ((unsigned*)lo)[0] = (unsigned)l0 | ((unsigned)l1 << 16);
  ((unsigned*)lo)[1] = (unsigned)l2 | ((unsigned)l3 << 16);
}

// A fragment: 16x32 bf16. lane m=l&15 holds row m; K interleave per ISA table.
__device__ __forceinline__ v16bf load_frag_a(const unsigned short* base, int lane) {
  int m = lane & 15, g = (lane >> 4) << 3;   // upper half-wave: K += 8
  FragU f;
  const unsigned short* row = base + m * ASTR;
#pragma unroll
  for (int d = 0; d < 8; ++d) {
    int K0 = (d < 4) ? (2*d + g) : (16 + 2*(d-4) + g);
    f.u[d] = *(const unsigned int*)(row + K0);
  }
  return f.v;
}

// B fragment: 32x16 bf16, stored channel-major in LDS ([ch][WSTR] = W row-major),
// so each dword is two consecutive-K values of one output channel.
__device__ __forceinline__ v16bf load_frag_b(const unsigned short* base, int lane) {
  int n = lane & 15, g = (lane >> 4) << 3;   // upper half-wave: K += 8
  FragU f;
  const unsigned short* row = base + n * WSTR;
#pragma unroll
  for (int d = 0; d < 8; ++d) f.u[d] = *(const unsigned int*)(row + ((d < 4) ? (2*d + g) : (16 + 2*(d-4) + g)));
  return f.v;
}

#define WMMA_BF16(A, B, Cacc) \
  __builtin_amdgcn_wmma_f32_16x16x32_bf16(false, (A), false, (B), (short)0, (Cacc), false, false)

__global__ void __launch_bounds__(256)
attn_pnts_kernel(const float* __restrict__ fea_center,
                 const float* __restrict__ fea_near,
                 const float* __restrict__ pwq,
                 const float* __restrict__ pwk,
                 const float* __restrict__ pwv,
                 float* __restrict__ out)
{
  extern __shared__ char smem[];
  unsigned short* aHi = (unsigned short*)(smem + OFF_AHI);
  unsigned short* aLo = (unsigned short*)(smem + OFF_ALO);
  unsigned short* wHi = (unsigned short*)(smem + OFF_WHI);
  unsigned short* wLo = (unsigned short*)(smem + OFF_WLO);
  float* stgF   = (float*)(smem + OFF_OUT);   // f32 staging (17408 floats) before GEMM
  float* kqOut  = (float*)(smem + OFF_OUT);   // [NROW][KQSTR]
  float* qOut   = (float*)(smem + OFF_QO);    // [16][KQSTR]
  float* vOut   = (float*)(smem + OFF_OUT);   // [NROW][VSTR] (reused after softmax)
  float* logits = (float*)(smem + OFF_LOG);   // [TP][17]
  float* wgt    = logits + TP*17;             // [TP][17]

  const int tid  = threadIdx.x;
  const int lane = tid & 31;
  const int wave = tid >> 5;
  const int b    = blockIdx.x >> 9;           // 512 blocks per batch
  const int n0   = (blockIdx.x & 511) * TP;

  // Per-workgroup feature tiles are fully contiguous in memory:
  const float* nearBase = fea_near   + ((size_t)b*NPTS + n0)*KNB*C;  // 16384 f32
  const float* centBase = fea_center + ((size_t)b*NPTS + n0)*C;      // 1024  f32

#if HAVE_ASYNC_LDS
  // ---- Phase 0a: kick off async DMA (global -> LDS staging), ASYNCcnt tracked ----
  for (int i = tid; i < 4096; i += 256)
    __builtin_amdgcn_global_load_async_to_lds_b128(
        (gv4i_p)(nearBase + i*4), (lv4i_p)(stgF + i*4), 0, 0);
  __builtin_amdgcn_global_load_async_to_lds_b128(
      (gv4i_p)(centBase + tid*4), (lv4i_p)(stgF + 16384 + tid*4), 0, 0);
#endif

  // ---- Phase 0b: stage weights as bf16 hi/lo (overlaps the feature DMA) ----
  // Combined channel map o: [0,64)=wk | [64,192)=wv | [192,256)=wq ; 8192 float4s
  for (int idx4 = tid; idx4 < 8192; idx4 += 256) {
    int o = idx4 >> 5, k = (idx4 << 2) & 127;
    float4 w4;
    if (o < 64)       w4 = ((const float4*)pwk)[idx4];
    else if (o < 192) w4 = ((const float4*)pwv)[idx4 - 2048];
    else              w4 = ((const float4*)pwq)[idx4 - 6144];
    split_store(wHi + o*WSTR + k, wLo + o*WSTR + k, w4);
  }

#if HAVE_ASYNC_LDS
  __builtin_amdgcn_s_wait_asynccnt(0);
  __syncthreads();   // all waves' DMA visible before conversion
#endif

  // ---- Phase 0c: convert features to bf16 hi/lo (from LDS staging or global) ----
  for (int idx4 = tid; idx4 < 4608; idx4 += 256) {
    int r, k;
    float4 v;
    if (idx4 < 4096) {                       // near rows 0..127 (contiguous)
      r = idx4 >> 5; k = (idx4 << 2) & 127;
#if HAVE_ASYNC_LDS
      v = ((const float4*)stgF)[idx4];
#else
      v = ((const float4*)nearBase)[idx4];
#endif
    } else if (idx4 < 4352) {                // center rows 128..135
      int t = idx4 - 4096; r = 128 + (t >> 5); k = (t << 2) & 127;
#if HAVE_ASYNC_LDS
      v = ((const float4*)stgF)[4096 + t];
#else
      v = ((const float4*)centBase)[t];
#endif
    } else {                                 // zero-pad rows 136..143
      int t = idx4 - 4352; r = 136 + (t >> 5); k = (t << 2) & 127;
      v = make_float4(0.f, 0.f, 0.f, 0.f);
    }
    split_store(aHi + r*ASTR + k, aLo + r*ASTR + k, v);
  }
  __syncthreads();

  // ---- Phase 1: k and q projections (bf16x3 split GEMM) ----
  // 40 tiles: t<36 -> k (tr=t/4, ch=(t%4)*16); t>=36 -> q (tr=8, ch=192+(t-36)*16)
  for (int i = 0; i < 5; ++i) {
    int t = wave + 8*i;
    int tr, chBase, isQ;
    if (t < 36) { tr = t >> 2; chBase = (t & 3) * 16; isQ = 0; }
    else        { tr = 8;      chBase = 192 + (t - 36) * 16; isQ = 1; }
    v8f acc = {0.f,0.f,0.f,0.f,0.f,0.f,0.f,0.f};
#pragma unroll
    for (int kk = 0; kk < 4; ++kk) {
      int kb = kk * 32;
      v16bf ah = load_frag_a(aHi + tr*16*ASTR + kb, lane);
      v16bf al = load_frag_a(aLo + tr*16*ASTR + kb, lane);
      v16bf bh = load_frag_b(wHi + chBase*WSTR + kb, lane);
      v16bf bl = load_frag_b(wLo + chBase*WSTR + kb, lane);
      acc = WMMA_BF16(ah, bh, acc);
      acc = WMMA_BF16(ah, bl, acc);
      acc = WMMA_BF16(al, bh, acc);
    }
    int n = lane & 15, half = lane >> 4;
#pragma unroll
    for (int r = 0; r < 8; ++r) {
      int M = r + 8*half;
      if (isQ) qOut[M*KQSTR + (chBase - 192) + n] = acc[r];
      else     kqOut[(tr*16 + M)*KQSTR + chBase + n] = acc[r];
    }
  }
  __syncthreads();

  // ---- q.k dots (17 logits per point) ----
  if (tid < TP*17) {
    int p = tid / 17, jj = tid % 17;
    int row = (jj < 16) ? (p*16 + jj) : (TP*KNB + p);
    float s = 0.f;
    for (int c0 = 0; c0 < CH; ++c0) s += qOut[p*KQSTR + c0] * kqOut[row*KQSTR + c0];
    logits[p*17 + jj] = s * 0.125f;  // 1/sqrt(64)
  }
  __syncthreads();

  // ---- softmax per point ----
  if (tid < TP) {
    float mx = -1e30f;
    for (int jj = 0; jj < 17; ++jj) mx = fmaxf(mx, logits[tid*17 + jj]);
    float e[17], sum = 0.f;
    for (int jj = 0; jj < 17; ++jj) { e[jj] = __expf(logits[tid*17 + jj] - mx); sum += e[jj]; }
    float inv = 1.f / sum;
    for (int jj = 0; jj < 17; ++jj) wgt[tid*17 + jj] = e[jj] * inv;
  }
  __syncthreads();  // also fences k/q reads before vOut overwrites the region

  // ---- Phase 2: v projection; each wave owns one 16-channel column, B hoisted ----
  {
    int chBase = 64 + wave * 16;
    v16bf bh[4], bl[4];
#pragma unroll
    for (int kk = 0; kk < 4; ++kk) {
      bh[kk] = load_frag_b(wHi + chBase*WSTR + kk*32, lane);
      bl[kk] = load_frag_b(wLo + chBase*WSTR + kk*32, lane);
    }
    for (int tr = 0; tr < NRT; ++tr) {
      v8f acc = {0.f,0.f,0.f,0.f,0.f,0.f,0.f,0.f};
#pragma unroll
      for (int kk = 0; kk < 4; ++kk) {
        v16bf ah = load_frag_a(aHi + tr*16*ASTR + kk*32, lane);
        v16bf al = load_frag_a(aLo + tr*16*ASTR + kk*32, lane);
        acc = WMMA_BF16(ah, bh[kk], acc);
        acc = WMMA_BF16(ah, bl[kk], acc);
        acc = WMMA_BF16(al, bh[kk], acc);
      }
      int n = lane & 15, half = lane >> 4;
#pragma unroll
      for (int r = 0; r < 8; ++r) {
        int M = r + 8*half;
        vOut[(tr*16 + M)*VSTR + (chBase - 64) + n] = acc[r];
      }
    }
  }
  __syncthreads();

  // ---- Phase 3: weighted combine + store ----
  for (int idx = tid; idx < TP * C; idx += 256) {
    int p = idx >> 7, ch = idx & 127;
    const float* wrow = wgt + p*17;
    float s = 0.f;
#pragma unroll 4
    for (int j = 0; j < 16; ++j) s += wrow[j] * vOut[(p*16 + j)*VSTR + ch];
    s += wrow[16] * vOut[(TP*KNB + p)*VSTR + ch];
    out[((size_t)b*NPTS + n0 + p)*C + ch] = s;
  }
}

extern "C" void kernel_launch(void* const* d_in, const int* in_sizes, int n_in,
                              void* d_out, int out_size, void* d_ws, size_t ws_size,
                              hipStream_t stream) {
  const float* fea_center = (const float*)d_in[0];
  const float* fea_near   = (const float*)d_in[1];
  const float* wq         = (const float*)d_in[2];
  const float* wk         = (const float*)d_in[3];
  const float* wv         = (const float*)d_in[4];
  float* out = (float*)d_out;

  dim3 grid(BS * (NPTS / TP));   // 4096 workgroups
  dim3 block(256);               // 8 wave32
  attn_pnts_kernel<<<grid, block, SMEM_BYTES, stream>>>(fea_center, fea_near,
                                                        wq, wk, wv, out);
}
